// GINEDecoder_89644557402627
// MI455X (gfx1250) — matrix operand order repaired and checked by reading, not compiled, not run on hardware
//
#include <hip/hip_runtime.h>

// ---------------------------------------------------------------------------
// GINE decoder for MI455X (gfx1250, wave32). All MLP matmuls run on
// v_wmma_f32_16x16x32_f16 (f32 accumulate). Pipeline:
//   1) edge encoder  MLP[16->64->64->64]  -> e  (workspace, f32)
//   2) node proj     MLP[128->64->64->64] -> x0
//   3) 3x GINE layer: zero agg; scatter relu(x[src]+e) into agg[dst];
//                     x = MLP[64->64->64->64](x + agg)
//   4) out proj      MLP[64->64->64->64]  -> d_out
// ---------------------------------------------------------------------------

typedef __attribute__((ext_vector_type(16))) _Float16 v16h;
typedef __attribute__((ext_vector_type(8)))  _Float16 v8h;
typedef __attribute__((ext_vector_type(8)))  float    v8f;

union HFrag { v16h v; v8h h[2]; };

__device__ __forceinline__ v8f wmma_f32_f16(v16h a, v16h b, v8f c) {
  // (neg_a, A, neg_b, B, c_mod, C, reuse_a, reuse_b)
  return __builtin_amdgcn_wmma_f32_16x16x32_f16(false, a, false, b, (short)0, c,
                                                false, false);
}

// A fragment (16x32 f16) from a row-major LDS tile [16][S] halves.
// CDNA5 layout: lane holds row = lane&15; half = lane>>4 selects K 0-7 / 8-15
// within each 16-K group; VGPRs 0-3 = group 0 (K base +0), 4-7 = group 1 (+16).
__device__ __forceinline__ v16h lds_afrag(const _Float16* act, int S, int ko, int lane) {
  const int row = lane & 15, hf = lane >> 4;
  const _Float16* p = act + row * S + ko + hf * 8;
  HFrag f;
  f.h[0] = *(const v8h*)(p);        // K = ko + hf*8 .. +7
  f.h[1] = *(const v8h*)(p + 16);   // K = ko + 16 + hf*8 .. +7
  return f.v;
}

// B fragment (32x16 f16) from LDS weights stored N-major: wt[n*KS + k].
// Lane holds column N = lane&15; half = lane>>4 selects K 0-7 / 8-15 within each 16-K group.
__device__ __forceinline__ v16h lds_bfrag(const _Float16* wt, int KS, int ko,
                                          int ntile, int lane) {
  const int col = (lane & 15) + ntile * 16, hf = lane >> 4;
  const _Float16* p = wt + col * KS + ko + hf * 8;
  HFrag f;
  f.h[0] = *(const v8h*)(p);        // K = ko + hf*8 .. +7  (VGPRs 0-3)
  f.h[1] = *(const v8h*)(p + 16);   // K = ko + 16 + hf*8 .. +7 (VGPRs 4-7)
  return f.v;
}

// Cooperatively stage W [K][64] f32 (row-major) into LDS as f16, transposed to
// N-major [64][KS]; rows K..KS-1 zero-padded so padded WMMA K-steps are clean.
__device__ __forceinline__ void stage_wt(const float* __restrict__ W, _Float16* wt,
                                         int K, int KS, int tid, int nth) {
  const int total = 64 * KS;
  for (int i = tid; i < total; i += nth) {
    const int n = i / KS, k = i % KS;
    wt[i] = (k < K) ? (_Float16)W[k * 64 + n] : (_Float16)0.0f;
  }
}

// One linear layer on a 16-row tile: act(16xK) * W(Kx64) + b, ReLU, result
// written back into the same per-wave LDS tile as f16 (for the next layer).
template <int K, int KS>
__device__ __forceinline__ void mlp_layer_tile(_Float16* act, int S,
                                               const _Float16* wt,
                                               const float* bias, int lane) {
  const v8f vz = {0.f, 0.f, 0.f, 0.f, 0.f, 0.f, 0.f, 0.f};
  v8f acc[4] = {vz, vz, vz, vz};
#pragma unroll
  for (int ko = 0; ko < K; ko += 32) {
    const v16h a = lds_afrag(act, S, ko, lane);
#pragma unroll
    for (int t = 0; t < 4; ++t)
      acc[t] = wmma_f32_f16(a, lds_bfrag(wt, KS, ko, t, lane), acc[t]);
  }
  const int hf = lane >> 4, c0 = lane & 15;
#pragma unroll
  for (int t = 0; t < 4; ++t) {
    const int col = c0 + t * 16;
    const float bv = bias[col];
#pragma unroll
    for (int r = 0; r < 8; ++r) {
      float v = acc[t][r] + bv;
      v = fmaxf(v, 0.0f);
      act[(hf * 8 + r) * S + col] = (_Float16)v;   // M = 8*half + r, N = col
    }
  }
}

// Final linear layer of an MLP (no ReLU): result -> global f32 [16][64] tile.
template <int K, int KS>
__device__ __forceinline__ void mlp_layer_global(const _Float16* act, int S,
                                                 const _Float16* wt,
                                                 const float* bias,
                                                 float* __restrict__ out, int lane) {
  const v8f vz = {0.f, 0.f, 0.f, 0.f, 0.f, 0.f, 0.f, 0.f};
  v8f acc[4] = {vz, vz, vz, vz};
#pragma unroll
  for (int ko = 0; ko < K; ko += 32) {
    const v16h a = lds_afrag(act, S, ko, lane);
#pragma unroll
    for (int t = 0; t < 4; ++t)
      acc[t] = wmma_f32_f16(a, lds_bfrag(wt, KS, ko, t, lane), acc[t]);
  }
  const int hf = lane >> 4, c0 = lane & 15;
#pragma unroll
  for (int t = 0; t < 4; ++t) {
    const int col = c0 + t * 16;
    const float bv = bias[col];
#pragma unroll
    for (int r = 0; r < 8; ++r)
      out[(hf * 8 + r) * 64 + col] = acc[t][r] + bv;
  }
}

// ---------------------------------------------------------------------------
// Edge encoder: MLP [16 -> 64 -> 64 -> 64] over 16-edge tiles, 8 waves/block.
// ---------------------------------------------------------------------------
__global__ __launch_bounds__(256) void k_edge_mlp(
    const float* __restrict__ ef, const float* __restrict__ W1,
    const float* __restrict__ b1, const float* __restrict__ W2,
    const float* __restrict__ b2, const float* __restrict__ W3,
    const float* __restrict__ b3, float* __restrict__ eout, int ntiles) {
  constexpr int S = 72;  // halves; 144 B row stride (16B-aligned)
  __shared__ _Float16 wt1[64 * 32];
  __shared__ _Float16 wt2[64 * 64];
  __shared__ _Float16 wt3[64 * 64];
  __shared__ float    bias[3][64];
  __shared__ _Float16 act_all[8][16 * S];

  const int tid = threadIdx.x, nth = blockDim.x;
  stage_wt(W1, wt1, 16, 32, tid, nth);   // K padded 16 -> 32, pad zeroed
  stage_wt(W2, wt2, 64, 64, tid, nth);
  stage_wt(W3, wt3, 64, 64, tid, nth);
  for (int i = tid; i < 64; i += nth) {
    bias[0][i] = b1[i]; bias[1][i] = b2[i]; bias[2][i] = b3[i];
  }
  __syncthreads();

  const int wave = tid >> 5, lane = tid & 31;
  const int tile = blockIdx.x * 8 + wave;
  if (tile >= ntiles) return;
  _Float16* act = act_all[wave];

  {  // stage 16x16 input (f32->f16), zero-pad K 16..31
    const int row = lane & 15, hf = lane >> 4;
    const float* ip = ef + ((long long)tile * 16 + row) * 16 + hf * 8;
    const float4 u0 = *(const float4*)(ip);
    const float4 u1 = *(const float4*)(ip + 4);
    v8h h;
    h[0] = (_Float16)u0.x; h[1] = (_Float16)u0.y; h[2] = (_Float16)u0.z; h[3] = (_Float16)u0.w;
    h[4] = (_Float16)u1.x; h[5] = (_Float16)u1.y; h[6] = (_Float16)u1.z; h[7] = (_Float16)u1.w;
    *(v8h*)(act + row * S + hf * 8) = h;
    const v8h z = {0, 0, 0, 0, 0, 0, 0, 0};
    *(v8h*)(act + row * S + 16 + hf * 8) = z;
  }
  mlp_layer_tile<32, 32>(act, S, wt1, bias[0], lane);
  mlp_layer_tile<64, 64>(act, S, wt2, bias[1], lane);
  mlp_layer_global<64, 64>(act, S, wt3, bias[2],
                           eout + (long long)tile * 16 * 64, lane);
}

// ---------------------------------------------------------------------------
// Generic 3-layer MLP over 16-row tiles (K_IN -> 64 -> 64 -> 64), optional
// fused elementwise add of a second input (for x + agg in the GINE update).
// ---------------------------------------------------------------------------
template <int K_IN, int WAVES>
__global__ __launch_bounds__(WAVES * 32) void k_mlp3(
    const float* __restrict__ in, const float* __restrict__ add_in,
    const float* __restrict__ W1, const float* __restrict__ b1,
    const float* __restrict__ W2, const float* __restrict__ b2,
    const float* __restrict__ W3, const float* __restrict__ b3,
    float* __restrict__ out, int ntiles) {
  constexpr int S = K_IN + 8;  // stride in halves, multiple of 8 -> 16B rows
  __shared__ _Float16 wt1[64 * K_IN];
  __shared__ _Float16 wt2[64 * 64];
  __shared__ _Float16 wt3[64 * 64];
  __shared__ float    bias[3][64];
  __shared__ _Float16 act_all[WAVES][16 * S];

  const int tid = threadIdx.x, nth = blockDim.x;
  stage_wt(W1, wt1, K_IN, K_IN, tid, nth);
  stage_wt(W2, wt2, 64, 64, tid, nth);
  stage_wt(W3, wt3, 64, 64, tid, nth);
  for (int i = tid; i < 64; i += nth) {
    bias[0][i] = b1[i]; bias[1][i] = b2[i]; bias[2][i] = b3[i];
  }
  __syncthreads();

  const int wave = tid >> 5, lane = tid & 31;
  const int tile = blockIdx.x * WAVES + wave;
  if (tile >= ntiles) return;
  _Float16* act = act_all[wave];

  {  // stage input tile (+ optional add), f32 -> f16
    const int row = lane & 15, hf = lane >> 4;
    const long long roff = ((long long)tile * 16 + row) * K_IN + hf * (K_IN / 2);
    const float* ip = in + roff;
    const float* ap = add_in ? (add_in + roff) : nullptr;
    _Float16* op = act + row * S + hf * (K_IN / 2);
#pragma unroll
    for (int c = 0; c < K_IN / 2; c += 8) {
      float4 u0 = *(const float4*)(ip + c);
      float4 u1 = *(const float4*)(ip + c + 4);
      if (ap) {
        const float4 a0 = *(const float4*)(ap + c);
        const float4 a1 = *(const float4*)(ap + c + 4);
        u0.x += a0.x; u0.y += a0.y; u0.z += a0.z; u0.w += a0.w;
        u1.x += a1.x; u1.y += a1.y; u1.z += a1.z; u1.w += a1.w;
      }
      v8h h;
      h[0] = (_Float16)u0.x; h[1] = (_Float16)u0.y; h[2] = (_Float16)u0.z; h[3] = (_Float16)u0.w;
      h[4] = (_Float16)u1.x; h[5] = (_Float16)u1.y; h[6] = (_Float16)u1.z; h[7] = (_Float16)u1.w;
      *(v8h*)(op + c) = h;
    }
  }
  mlp_layer_tile<K_IN, K_IN>(act, S, wt1, bias[0], lane);
  mlp_layer_tile<64, 64>(act, S, wt2, bias[1], lane);
  mlp_layer_global<64, 64>(act, S, wt3, bias[2],
                           out + (long long)tile * 16 * 64, lane);
}

// ---------------------------------------------------------------------------
// Message + scatter: agg[dst] += relu(x[src] + e), 4 threads per edge,
// float4 loads, global_atomic_add_f32 scatter.
// ---------------------------------------------------------------------------
__global__ __launch_bounds__(256) void k_msg_scatter(
    const float* __restrict__ x, const float* __restrict__ e,
    const int* __restrict__ src, const int* __restrict__ dst,
    float* __restrict__ agg, int E) {
  const long long t = (long long)blockIdx.x * blockDim.x + threadIdx.x;
  const long long edge = t >> 2;
  if (edge >= (long long)E) return;
  const int part = (int)(t & 3) * 16;
  const int s = src[edge], d = dst[edge];
  const float4* xp = (const float4*)(x + (long long)s * 64 + part);
  const float4* ep = (const float4*)(e + edge * 64 + part);
  float* ap = agg + (long long)d * 64 + part;
#pragma unroll
  for (int j = 0; j < 4; ++j) {
    const float4 xv = xp[j];
    const float4 ev = ep[j];
    atomicAdd(ap + j * 4 + 0, fmaxf(xv.x + ev.x, 0.0f));
    atomicAdd(ap + j * 4 + 1, fmaxf(xv.y + ev.y, 0.0f));
    atomicAdd(ap + j * 4 + 2, fmaxf(xv.z + ev.z, 0.0f));
    atomicAdd(ap + j * 4 + 3, fmaxf(xv.w + ev.w, 0.0f));
  }
}

__global__ __launch_bounds__(256) void k_zero4(float* __restrict__ p, int n4) {
  const int i = blockIdx.x * blockDim.x + threadIdx.x;
  if (i < n4) ((float4*)p)[i] = make_float4(0.f, 0.f, 0.f, 0.f);
}

// ---------------------------------------------------------------------------
extern "C" void kernel_launch(void* const* d_in, const int* in_sizes, int n_in,
                              void* d_out, int out_size, void* d_ws, size_t ws_size,
                              hipStream_t stream) {
  (void)out_size; (void)ws_size;
  const float* node_feat = (const float*)d_in[0];
  const float* edge_feat = (const float*)d_in[1];
  const int*   src       = (const int*)d_in[2];
  const int*   dst       = (const int*)d_in[3];
  const int N = in_sizes[0] / 128;
  const int E = in_sizes[1] / 16;

  // Param leaves: (W,b) x3 per MLP. Handle both dict-insertion order
  // (edge_encoder, node_proj, gine, out_proj) and JAX sorted-key pytree order
  // (edge_encoder, gine, node_proj, out_proj) — discriminated by node_proj's
  // first weight being 128*64 elements.
  const float* ee[6]; const float* npj[6]; const float* gi[3][6]; const float* opj[6];
  int idx = 4;
  for (int i = 0; i < 6; ++i) ee[i] = (const float*)d_in[idx++];
  const bool insertion = (idx < n_in) && (in_sizes[idx] == 128 * 64);
  if (insertion) {
    for (int i = 0; i < 6; ++i) npj[i] = (const float*)d_in[idx++];
    for (int l = 0; l < 3; ++l)
      for (int i = 0; i < 6; ++i) gi[l][i] = (const float*)d_in[idx++];
  } else {
    for (int l = 0; l < 3; ++l)
      for (int i = 0; i < 6; ++i) gi[l][i] = (const float*)d_in[idx++];
    for (int i = 0; i < 6; ++i) npj[i] = (const float*)d_in[idx++];
  }
  for (int i = 0; i < 6; ++i) opj[i] = (const float*)d_in[idx++];

  // Workspace layout (f32): e [E,64] | x0 [N,64] | x1 [N,64] | agg [N,64]
  float* e_buf = (float*)d_ws;
  float* x0    = e_buf + (size_t)E * 64;
  float* x1    = x0 + (size_t)N * 64;
  float* agg   = x1 + (size_t)N * 64;

  const int Etiles = (E + 15) / 16;
  const int Ntiles = (N + 15) / 16;

  k_edge_mlp<<<dim3((Etiles + 7) / 8), dim3(256), 0, stream>>>(
      edge_feat, ee[0], ee[1], ee[2], ee[3], ee[4], ee[5], e_buf, Etiles);

  k_mlp3<128, 4><<<dim3((Ntiles + 3) / 4), dim3(128), 0, stream>>>(
      node_feat, nullptr, npj[0], npj[1], npj[2], npj[3], npj[4], npj[5], x0, Ntiles);

  float* xc = x0; float* xn = x1;
  for (int l = 0; l < 3; ++l) {
    const int n4 = N * 16;  // N*64/4 float4s
    k_zero4<<<dim3((n4 + 255) / 256), dim3(256), 0, stream>>>(agg, n4);
    const long long tot = (long long)E * 4;
    k_msg_scatter<<<dim3((unsigned)((tot + 255) / 256)), dim3(256), 0, stream>>>(
        xc, e_buf, src, dst, agg, E);
    k_mlp3<64, 8><<<dim3((Ntiles + 7) / 8), dim3(256), 0, stream>>>(
        xc, agg, gi[l][0], gi[l][1], gi[l][2], gi[l][3], gi[l][4], gi[l][5], xn, Ntiles);
    float* tmp = xc; xc = xn; xn = tmp;
  }
  k_mlp3<64, 8><<<dim3((Ntiles + 7) / 8), dim3(256), 0, stream>>>(
      xc, nullptr, opj[0], opj[1], opj[2], opj[3], opj[4], opj[5],
      (float*)d_out, Ntiles);
}
